// DecoderModel_61641370632323
// MI455X (gfx1250) — compile-verified, matrix operand order checked by reading, not run on hardware
//
#include <hip/hip_runtime.h>
#include <hip/hip_bf16.h>
#include <stdint.h>

// ---------------- problem constants ----------------
constexpr int L_  = 4;
constexpr int S_  = 1024;
constexpr int H_  = 2048;
constexpr int NH_ = 16;
constexpr int HD_ = 128;
constexpr int V_  = 32000;
constexpr int P_  = 5632;

typedef __attribute__((ext_vector_type(16))) _Float16 v16h;
typedef __attribute__((ext_vector_type(8)))  float    v8f;

union U16 { v16h v; unsigned u[8]; };
union H4  { _Float16 h[4]; unsigned long long ull; };

// K-offset of VGPR j for a 16-bit A/B fragment (16x32), lane-half kh (ISA 7.12.2)
__device__ __forceinline__ int kidx(int j, int kh) {
    return (j < 4) ? (2 * j + 8 * kh) : (16 + 2 * (j - 4) + 8 * kh);
}

__device__ __forceinline__ v8f v8f_zero() {
    v8f z = {0.f, 0.f, 0.f, 0.f, 0.f, 0.f, 0.f, 0.f};
    return z;
}

// ---------------- RMSNorm: fp32 in -> f16 out (+optional fp32 out) ----------------
__global__ void rmsnorm_kernel(const float* __restrict__ x, const float* __restrict__ w,
                               _Float16* __restrict__ oh, float* __restrict__ of, int cols) {
    int row = blockIdx.x;
    const float* xr = x + (size_t)row * cols;
    __shared__ float red[8];
    __shared__ float rtot;
    float s = 0.f;
    for (int c = threadIdx.x; c < cols; c += 256) { float v = xr[c]; s += v * v; }
#pragma unroll
    for (int m = 16; m >= 1; m >>= 1) s += __shfl_xor(s, m, 32);
    if ((threadIdx.x & 31) == 0) red[threadIdx.x >> 5] = s;
    __syncthreads();
    if (threadIdx.x == 0) {
        float t = 0.f;
#pragma unroll
        for (int k = 0; k < 8; k++) t += red[k];
        rtot = rsqrtf(t / (float)cols + 1e-5f);
    }
    __syncthreads();
    float rs = rtot;
    for (int c = threadIdx.x; c < cols; c += 256) {
        float v = xr[c] * rs * w[c];
        oh[(size_t)row * cols + c] = (_Float16)v;
        if (of) of[(size_t)row * cols + c] = v;
    }
}

// ---------------- WMMA GEMM: C[M,N] = A(f16)[M,K] * W(f32)[N,K]^T (+resid) ----------------
__global__ __launch_bounds__(128) void gemm_f16_wmma(
        const _Float16* __restrict__ A, const float* __restrict__ W,
        const float* __restrict__ resid, float* __restrict__ C,
        int M, int N, int K) {
    __shared__ _Float16 wt[2][64 * 36];
    const int wave = threadIdx.x >> 5;
    const int lane = threadIdx.x & 31;
    const int lh = lane & 15, kh = lane >> 4;
    const int m0 = blockIdx.x * 64 + wave * 16;
    const int n0 = blockIdx.y * 64;

    v8f acc[4];
#pragma unroll
    for (int s = 0; s < 4; s++) acc[s] = v8f_zero();

    auto stage = [&](int k0, int b) {
#pragma unroll
        for (int i = 0; i < 4; i++) {
            int gi = threadIdx.x + 128 * i;
            int r = gi >> 3, cg = gi & 7;
            float4 wv = *(const float4*)&W[(size_t)(n0 + r) * K + k0 + cg * 4];
            H4 p;
            p.h[0] = (_Float16)wv.x; p.h[1] = (_Float16)wv.y;
            p.h[2] = (_Float16)wv.z; p.h[3] = (_Float16)wv.w;
            *(unsigned long long*)&wt[b][r * 36 + cg * 4] = p.ull;
        }
    };

    stage(0, 0);
    int p = 0;
    for (int k0 = 0; k0 < K; k0 += 32) {
        __syncthreads();
        if (k0 + 32 < K) stage(k0 + 32, p ^ 1);

        U16 af;
        const _Float16* Ar = A + (size_t)(m0 + lh) * K + k0;
#pragma unroll
        for (int j = 0; j < 8; j++)
            af.u[j] = *(const unsigned*)(Ar + kidx(j, kh));

#pragma unroll
        for (int s = 0; s < 4; s++) {
            U16 bf;
            const _Float16* Br = &wt[p][(s * 16 + lh) * 36];
#pragma unroll
            for (int j = 0; j < 8; j++)
                bf.u[j] = *(const unsigned*)(Br + kidx(j, kh));
            acc[s] = __builtin_amdgcn_wmma_f32_16x16x32_f16(false, af.v, false, bf.v,
                                                            (short)0, acc[s], false, false);
        }
        p ^= 1;
    }

#pragma unroll
    for (int s = 0; s < 4; s++) {
        int col = n0 + s * 16 + lh;
#pragma unroll
        for (int r = 0; r < 8; r++) {
            int row = m0 + 8 * kh + r;
            size_t idx = (size_t)row * N + col;
            float v = acc[s][r];
            if (resid) v += resid[idx];
            C[idx] = v;
        }
    }
}

// ---------------- RoPE (interleaved) fp32 [S,NH,HD] -> f16 [NH][S][HD] ----------------
__global__ void rope_kernel(const float* __restrict__ xf, const float* __restrict__ cosp,
                            const float* __restrict__ sinp, _Float16* __restrict__ out) {
    int i = blockIdx.x * 256 + threadIdx.x;
    int d2 = i & (HD_ / 2 - 1);
    int h  = (i >> 6) & (NH_ - 1);
    int s  = i >> 10;
    float x1 = xf[(size_t)s * H_ + h * HD_ + 2 * d2];
    float x2 = xf[(size_t)s * H_ + h * HD_ + 2 * d2 + 1];
    float c  = cosp[s * (HD_ / 2) + d2];
    float si = sinp[s * (HD_ / 2) + d2];
    size_t o = ((size_t)h * S_ + s) * HD_ + 2 * d2;
    out[o]     = (_Float16)(c * x1 - si * x2);
    out[o + 1] = (_Float16)(si * x1 + c * x2);
}

// ---------------- V convert+transpose: fp32 [S,NH,HD] -> f16 [NH][HD][S] ----------------
__global__ void convv_kernel(const float* __restrict__ vf, _Float16* __restrict__ out) {
    int i = blockIdx.x * 256 + threadIdx.x;
    int s = i & (S_ - 1);
    int d = (i >> 10) & (HD_ - 1);
    int h = i >> 17;
    out[i] = (_Float16)vf[(size_t)s * H_ + h * HD_ + d];
}

// ---------------- Flash attention: 1 wave per (head, 16-query tile) ----------------
// K/V tiles staged via CDNA5 async global->LDS (ASYNCcnt), double buffered.
__global__ __launch_bounds__(32) void attn_kernel(
        const _Float16* __restrict__ Qh, const _Float16* __restrict__ Kh,
        const _Float16* __restrict__ VTh, _Float16* __restrict__ Oh) {
    __shared__ _Float16 kt[2][32 * 136];   // 32 keys x 128 d, row stride 136 halves (272B)
    __shared__ _Float16 vt[2][128 * 40];   // 128 d x 32 keys, row stride 40 halves (80B)
    __shared__ _Float16 pb[16 * 34];
    int blk = blockIdx.x;
    int h  = blk & (NH_ - 1);
    int q0 = (blk >> 4) * 16;
    int lane = threadIdx.x & 31;
    int lh = lane & 15, kh = lane >> 4;

    // async-stage one 32-key K tile + V tile into buffer b
    auto issue_tiles = [&](int s0, int b) {
        const char* Kb = (const char*)(Kh + ((size_t)h * S_ + s0) * HD_);
        unsigned kbase = (unsigned)(size_t)&kt[b][0];
#pragma unroll
        for (int i = 0; i < 16; i++) {
            int id = i * 32 + lane;
            int r = id >> 4, c = id & 15;          // key row, 16B chunk
            const void* g = Kb + r * 256 + c * 16;
            unsigned l = kbase + r * 272 + c * 16;
            asm volatile("global_load_async_to_lds_b128 %0, %1, off"
                         :: "v"(l), "v"(g) : "memory");
        }
        const char* Vb = (const char*)(VTh + (size_t)h * HD_ * S_ + s0);
        unsigned vbase = (unsigned)(size_t)&vt[b][0];
#pragma unroll
        for (int i = 0; i < 16; i++) {
            int id = i * 32 + lane;
            int r = id >> 2, c = id & 3;           // d row, 16B chunk
            const void* g = Vb + (size_t)r * (S_ * 2) + c * 16;
            unsigned l = vbase + r * 80 + c * 16;
            asm volatile("global_load_async_to_lds_b128 %0, %1, off"
                         :: "v"(l), "v"(g) : "memory");
        }
    };

    v16h qfrag[4];
    const _Float16* Qb = Qh + ((size_t)h * S_ + q0 + lh) * HD_;
#pragma unroll
    for (int c = 0; c < 4; c++) {
        U16 u;
#pragma unroll
        for (int j = 0; j < 8; j++)
            u.u[j] = *(const unsigned*)(Qb + c * 32 + kidx(j, kh));
        qfrag[c] = u.v;
    }

    float m_i[8], l_i[8];
    v8f o[8];
#pragma unroll
    for (int r = 0; r < 8; r++) { m_i[r] = -1e30f; l_i[r] = 0.f; }
#pragma unroll
    for (int t = 0; t < 8; t++) o[t] = v8f_zero();

    const float scale = 0.08838834764831845f;   // 1/sqrt(128)
    int nkb = (q0 + 15) / 32 + 1;

    issue_tiles(0, 0);
    int buf = 0;

    for (int kb = 0; kb < nkb; kb++) {
        int s0 = kb * 32;
        if (kb + 1 < nkb) {
            asm volatile("s_wait_dscnt 0x0" ::: "memory");   // prior LDS reads done
            issue_tiles(s0 + 32, buf ^ 1);                   // prefetch next tiles
            asm volatile("s_wait_asynccnt 0x20" ::: "memory"); // current tiles landed
        } else {
            asm volatile("s_wait_asynccnt 0x0" ::: "memory");
        }

        v8f sc[2]; sc[0] = v8f_zero(); sc[1] = v8f_zero();
#pragma unroll
        for (int t = 0; t < 2; t++) {
#pragma unroll
            for (int c = 0; c < 4; c++) {
                U16 b;
                const _Float16* Kr = &kt[buf][(t * 16 + lh) * 136 + c * 32];
#pragma unroll
                for (int j = 0; j < 8; j++)
                    b.u[j] = *(const unsigned*)(Kr + kidx(j, kh));
                sc[t] = __builtin_amdgcn_wmma_f32_16x16x32_f16(false, qfrag[c], false, b.v,
                                                               (short)0, sc[t], false, false);
            }
        }

        float al[8];
#pragma unroll
        for (int r = 0; r < 8; r++) {
            int qrow = q0 + 8 * kh + r;
            float t0 = (s0 + lh      <= qrow) ? sc[0][r] * scale : -1e30f;
            float t1 = (s0 + 16 + lh <= qrow) ? sc[1][r] * scale : -1e30f;
            float mx = fmaxf(t0, t1);
#pragma unroll
            for (int m = 8; m >= 1; m >>= 1) mx = fmaxf(mx, __shfl_xor(mx, m, 32));
            float nm = fmaxf(m_i[r], mx);
            al[r] = __expf(m_i[r] - nm);
            float p0 = __expf(t0 - nm);
            float p1 = __expf(t1 - nm);
            pb[(8 * kh + r) * 34 + lh]      = (_Float16)p0;
            pb[(8 * kh + r) * 34 + 16 + lh] = (_Float16)p1;
            float ps = p0 + p1;
#pragma unroll
            for (int m = 8; m >= 1; m >>= 1) ps += __shfl_xor(ps, m, 32);
            l_i[r] = l_i[r] * al[r] + ps;
            m_i[r] = nm;
        }
        __syncthreads();

        U16 pf;
#pragma unroll
        for (int j = 0; j < 8; j++)
            pf.u[j] = *(const unsigned*)&pb[lh * 34 + kidx(j, kh)];

#pragma unroll
        for (int t = 0; t < 8; t++) {
#pragma unroll
            for (int r = 0; r < 8; r++) o[t][r] *= al[r];
            U16 vfg;
            const _Float16* Vr = &vt[buf][(t * 16 + lh) * 40];
#pragma unroll
            for (int j = 0; j < 8; j++)
                vfg.u[j] = *(const unsigned*)(Vr + kidx(j, kh));
            o[t] = __builtin_amdgcn_wmma_f32_16x16x32_f16(false, pf.v, false, vfg.v,
                                                          (short)0, o[t], false, false);
        }
        __syncthreads();
        buf ^= 1;
    }

#pragma unroll
    for (int t = 0; t < 8; t++) {
#pragma unroll
        for (int r = 0; r < 8; r++) {
            int srow = q0 + 8 * kh + r;
            Oh[(size_t)srow * H_ + h * HD_ + t * 16 + lh] = (_Float16)(o[t][r] / l_i[r]);
        }
    }
}

// ---------------- SwiGLU elementwise ----------------
__global__ void swiglu_kernel(const float* __restrict__ g, const float* __restrict__ u,
                              _Float16* __restrict__ t) {
    size_t i = (size_t)blockIdx.x * 256 + threadIdx.x;
    float gv = g[i];
    float sv = gv / (1.f + __expf(-gv));
    t[i] = (_Float16)(sv * u[i]);
}

// ---------------- Logits GEMV ----------------
__global__ void logits_kernel(const float* __restrict__ xf, const float* __restrict__ ow,
                              float* __restrict__ out) {
    int v = blockIdx.x * 8 + (threadIdx.x >> 5);
    int lane = threadIdx.x & 31;
    const float4* wr4 = (const float4*)(ow + (size_t)v * H_);
    const float4* x4  = (const float4*)xf;
    float s = 0.f;
#pragma unroll 4
    for (int d = lane; d < H_ / 4; d += 32) {
        float4 a = wr4[d], b = x4[d];
        s += a.x * b.x + a.y * b.y + a.z * b.z + a.w * b.w;
    }
#pragma unroll
    for (int m = 16; m >= 1; m >>= 1) s += __shfl_xor(s, m, 32);
    if (lane == 0) out[v] = s;
}

// ---------------- host orchestration ----------------
extern "C" void kernel_launch(void* const* d_in, const int* in_sizes, int n_in,
                              void* d_out, int out_size, void* d_ws, size_t ws_size,
                              hipStream_t stream) {
    const float* x_in   = (const float*)d_in[0];
    const float* cosp   = (const float*)d_in[2];
    const float* sinp   = (const float*)d_in[3];
    const float* norm1w = (const float*)d_in[6];
    const float* norm2w = (const float*)d_in[7];
    const float* wq     = (const float*)d_in[8];
    const float* wk     = (const float*)d_in[9];
    const float* wv     = (const float*)d_in[10];
    const float* wo     = (const float*)d_in[11];
    const float* w1     = (const float*)d_in[12];
    const float* w2     = (const float*)d_in[13];
    const float* w3     = (const float*)d_in[14];
    const float* fnw    = (const float*)d_in[15];
    const float* outw   = (const float*)d_in[16];
    float* logits = (float*)d_out;

    char* base = (char*)d_ws;
    size_t off = 0;
    auto alloc = [&](size_t bytes) { char* p = base + off; off = (off + bytes + 255) & ~(size_t)255; return p; };
    float*    xcur  = (float*)   alloc((size_t)S_ * H_ * 4);
    float*    hres  = (float*)   alloc((size_t)S_ * H_ * 4);
    float*    qf    = (float*)   alloc((size_t)S_ * H_ * 4);
    float*    kf    = (float*)   alloc((size_t)S_ * H_ * 4);
    float*    vf    = (float*)   alloc((size_t)S_ * H_ * 4);
    float*    gf    = (float*)   alloc((size_t)S_ * P_ * 4);
    float*    uf    = (float*)   alloc((size_t)S_ * P_ * 4);
    float*    xfin  = (float*)   alloc((size_t)H_ * 4);
    _Float16* xnh   = (_Float16*)alloc((size_t)S_ * H_ * 2);
    _Float16* qh    = (_Float16*)alloc((size_t)S_ * H_ * 2);
    _Float16* kh2   = (_Float16*)alloc((size_t)S_ * H_ * 2);
    _Float16* vth   = (_Float16*)alloc((size_t)S_ * H_ * 2);
    _Float16* atth  = (_Float16*)alloc((size_t)S_ * H_ * 2);
    _Float16* th    = (_Float16*)alloc((size_t)S_ * P_ * 2);
    (void)ws_size; (void)in_sizes; (void)n_in; (void)out_size;

    hipMemcpyAsync(xcur, x_in, (size_t)S_ * H_ * 4, hipMemcpyDeviceToDevice, stream);

    dim3 gHH(S_ / 64, H_ / 64);
    dim3 gHP(S_ / 64, P_ / 64);
    dim3 blk128(128);

    for (int i = 0; i < L_; i++) {
        const float* wq_i = wq + (size_t)i * H_ * H_;
        const float* wk_i = wk + (size_t)i * H_ * H_;
        const float* wv_i = wv + (size_t)i * H_ * H_;
        const float* wo_i = wo + (size_t)i * H_ * H_;
        const float* w1_i = w1 + (size_t)i * P_ * H_;
        const float* w2_i = w2 + (size_t)i * H_ * P_;
        const float* w3_i = w3 + (size_t)i * P_ * H_;

        rmsnorm_kernel<<<S_, 256, 0, stream>>>(xcur, norm1w + (size_t)i * H_, xnh, nullptr, H_);

        gemm_f16_wmma<<<gHH, blk128, 0, stream>>>(xnh, wq_i, nullptr, qf, S_, H_, H_);
        gemm_f16_wmma<<<gHH, blk128, 0, stream>>>(xnh, wk_i, nullptr, kf, S_, H_, H_);
        gemm_f16_wmma<<<gHH, blk128, 0, stream>>>(xnh, wv_i, nullptr, vf, S_, H_, H_);

        rope_kernel<<<(S_ * NH_ * HD_ / 2) / 256, 256, 0, stream>>>(qf, cosp, sinp, qh);
        rope_kernel<<<(S_ * NH_ * HD_ / 2) / 256, 256, 0, stream>>>(kf, cosp, sinp, kh2);
        convv_kernel<<<(NH_ * HD_ * S_) / 256, 256, 0, stream>>>(vf, vth);

        attn_kernel<<<NH_ * (S_ / 16), 32, 0, stream>>>(qh, kh2, vth, atth);

        gemm_f16_wmma<<<gHH, blk128, 0, stream>>>(atth, wo_i, xcur, hres, S_, H_, H_);

        rmsnorm_kernel<<<S_, 256, 0, stream>>>(hres, norm2w + (size_t)i * H_, xnh, nullptr, H_);

        gemm_f16_wmma<<<gHP, blk128, 0, stream>>>(xnh, w1_i, nullptr, gf, S_, P_, H_);
        gemm_f16_wmma<<<gHP, blk128, 0, stream>>>(xnh, w3_i, nullptr, uf, S_, P_, H_);

        swiglu_kernel<<<(S_ * P_) / 256, 256, 0, stream>>>(gf, uf, th);

        gemm_f16_wmma<<<gHH, blk128, 0, stream>>>(th, w2_i, hres, xcur, S_, H_, P_);
    }

    rmsnorm_kernel<<<1, 256, 0, stream>>>(xcur + (size_t)(S_ - 1) * H_, fnw, xnh, xfin, H_);
    logits_kernel<<<V_ / 8, 256, 0, stream>>>(xfin, outw, logits);
}
